// AdaptiveDecoder_71176198029447
// MI455X (gfx1250) — matrix operand order repaired and checked by reading, not compiled, not run on hardware
//
#include <hip/hip_runtime.h>

typedef _Float16 half_t;
typedef unsigned int u32;
typedef unsigned short u16;
typedef __attribute__((ext_vector_type(16))) _Float16 v16h;
typedef __attribute__((ext_vector_type(8)))  float    v8f;
typedef __attribute__((ext_vector_type(4)))  int      v4i;
typedef __attribute__((address_space(1))) v4i* g_v4i;
typedef __attribute__((address_space(3))) v4i* l_v4i;

union U16h { v16h v; u32 u[8]; half_t h[16]; };
union U8f  { v8f v; float f[8]; };
union H8   { uint4 q; half_t h[8]; };

__device__ __forceinline__ v8f wmma_f32_f16(v16h a, v16h b, v8f c) {
  return __builtin_amdgcn_wmma_f32_16x16x32_f16(false, a, false, b, (short)0, c, false, false);
}

#if __has_builtin(__builtin_amdgcn_global_load_async_to_lds_b128)
#define HAVE_ASYNC_LDS 1
#else
#define HAVE_ASYNC_LDS 0
#endif

#if __has_builtin(__builtin_amdgcn_s_wait_asynccnt)
#define WAIT_ASYNC() __builtin_amdgcn_s_wait_asynccnt(0)
#else
#define WAIT_ASYNC() asm volatile("s_wait_asynccnt 0x0" ::: "memory")
#endif

// ---------------- small utility kernels ----------------
__global__ void k_cvt_f16(const float* __restrict__ in, half_t* __restrict__ out, int n) {
  int i = blockIdx.x * 256 + threadIdx.x;
  if (i < n) out[i] = (half_t)in[i];
}

__global__ void k_cvt_f16_T(const float* __restrict__ in, half_t* __restrict__ out, int K, int D) {
  int i = blockIdx.x * 256 + threadIdx.x;
  if (i < K * D) { int k = i / D, d = i % D; out[d * K + k] = (half_t)in[i]; }
}

__global__ void k_init_feat(const float* __restrict__ gf, half_t* __restrict__ feat, int rows, int Nper) {
  int i = blockIdx.x * 256 + threadIdx.x;
  if (i >= rows * 64) return;
  int row = i >> 6; int c = (i & 63) << 3;
  int b = row / Nper;
  const float* s = gf + b * 512 + c;
  H8 o;
  #pragma unroll
  for (int j = 0; j < 8; ++j) o.h[j] = (half_t)s[j];
  *(uint4*)(feat + (size_t)row * 512 + c) = o.q;
}

__global__ void k_init_xyz(const float* __restrict__ ip, float* __restrict__ xyz, int n) {
  int i = blockIdx.x * 256 + threadIdx.x;
  if (i < n) xyz[i] = ip[i % 768];
}

__global__ void k_build_comb(const half_t* __restrict__ feat, const float* __restrict__ gf,
                             half_t* __restrict__ comb, int rows, int Nper) {
  int i = blockIdx.x * 256 + threadIdx.x;
  if (i >= rows * 128) return;
  int row = i >> 7; int c = (i & 127) << 3;
  if (c < 512) {
    *(uint4*)(comb + (size_t)row * 1024 + c) = *(const uint4*)(feat + (size_t)row * 512 + c);
  } else {
    int b = row / Nper;
    const float* s = gf + b * 512 + (c - 512);
    H8 o;
    #pragma unroll
    for (int j = 0; j < 8; ++j) o.h[j] = (half_t)s[j];
    *(uint4*)(comb + (size_t)row * 1024 + c) = o.q;
  }
}

__global__ void k_dup_feat(const half_t* __restrict__ fr, half_t* __restrict__ feat2, int rows2, int N2) {
  int i = blockIdx.x * 256 + threadIdx.x;
  if (i >= rows2 * 64) return;
  int row = i >> 6; int c = (i & 63) << 3;
  int b = row / N2; int n = row % N2; int N = N2 >> 1;
  int src = b * N + (n < N ? n : n - N);
  *(uint4*)(feat2 + (size_t)row * 512 + c) = *(const uint4*)(fr + (size_t)src * 512 + c);
}

// ---------------- WMMA GEMM: out[M,D] = X[M,K] @ W[K,D] + bias (+res), f16 in, f32 acc ----
// WG = 128 threads (4 waves, 2x2 wave grid), WG tile 128x128, 64x64 per wave.
// 16 individually named v8f accumulators (no arrays -> no SROA failure -> no scratch).
__global__ __launch_bounds__(128, 1) void k_gemm(
    const half_t* __restrict__ X, const half_t* __restrict__ W,
    const float* __restrict__ bias,
    float* __restrict__ outF, half_t* __restrict__ outH,
    const half_t* __restrict__ res,
    int M, int K, int D)
{
  __shared__ half_t Xs[128][40];
  __shared__ half_t Ws[128][40];   // transposed: Ws[col][k]
  int tid = threadIdx.x;
  int w = tid >> 5, lane = tid & 31;
  int n = lane & 15, hlf = lane >> 4;
  int rw = w >> 1, cw = w & 1;
  int row0 = blockIdx.y * 128, col0 = blockIdx.x * 128;

  v8f a00 = {}, a01 = {}, a02 = {}, a03 = {};
  v8f a10 = {}, a11 = {}, a12 = {}, a13 = {};
  v8f a20 = {}, a21 = {}, a22 = {}, a23 = {};
  v8f a30 = {}, a31 = {}, a32 = {}, a33 = {};

  int xr = tid;                            // one X row per thread (128 rows x 32 halves)
  int wk = tid >> 2, wc = (tid & 3) << 5;  // W: 32 k-rows x 128 cols, 32 halves/thread

  for (int k0 = 0; k0 < K; k0 += 32) {
    // ---- stage X tile: 128x32 f16 ----
    const half_t* gx = X + (size_t)(row0 + xr) * K + k0;
#if HAVE_ASYNC_LDS
    #pragma unroll
    for (int s = 0; s < 4; ++s)
      __builtin_amdgcn_global_load_async_to_lds_b128(
          (g_v4i)(gx + s * 8), (l_v4i)&Xs[xr][s * 8], 0, 0);
#else
    {
      const uint4* xs = (const uint4*)gx;
      uint4 x0 = xs[0], x1 = xs[1], x2 = xs[2], x3 = xs[3];
      *(uint4*)&Xs[xr][0]  = x0;
      *(uint4*)&Xs[xr][8]  = x1;
      *(uint4*)&Xs[xr][16] = x2;
      *(uint4*)&Xs[xr][24] = x3;
    }
#endif
    // ---- stage W tile transposed: Ws[col][k], 32 k-rows x 128 cols ----
    {
      const half_t* gw = W + (size_t)(k0 + wk) * D + col0 + wc;
      #pragma unroll
      for (int part = 0; part < 4; ++part) {
        H8 t; t.q = *(const uint4*)(gw + part * 8);
        #pragma unroll
        for (int j = 0; j < 8; ++j) Ws[wc + part * 8 + j][wk] = t.h[j];
      }
      if (k0 + 32 < K) __builtin_prefetch(gw + (size_t)32 * D, 0, 0);
    }
#if HAVE_ASYNC_LDS
    WAIT_ASYNC();
#endif
    __syncthreads();

    U16h bf0, bf1, bf2, bf3;
#define GEMM_LOADB(NT, BF)                                                    \
    { _Pragma("unroll")                                                       \
      for (int j = 0; j < 8; ++j) {                                           \
        int kb = (hlf << 4) + (j << 1);                                       \
        BF.u[j] = *(const u32*)&Ws[cw * 64 + NT * 16 + n][kb];                \
      } }
    GEMM_LOADB(0, bf0) GEMM_LOADB(1, bf1) GEMM_LOADB(2, bf2) GEMM_LOADB(3, bf3)

#define GEMM_ROW(MT, A0, A1, A2, A3)                                          \
    { U16h af;                                                                \
      _Pragma("unroll")                                                       \
      for (int j = 0; j < 8; ++j) {                                           \
        int ka = (hlf << 4) + (j << 1);                                       \
        af.u[j] = *(const u32*)&Xs[rw * 64 + MT * 16 + n][ka];                \
      }                                                                       \
      A0 = wmma_f32_f16(af.v, bf0.v, A0);                                     \
      A1 = wmma_f32_f16(af.v, bf1.v, A1);                                     \
      A2 = wmma_f32_f16(af.v, bf2.v, A2);                                     \
      A3 = wmma_f32_f16(af.v, bf3.v, A3); }
    GEMM_ROW(0, a00, a01, a02, a03)
    GEMM_ROW(1, a10, a11, a12, a13)
    GEMM_ROW(2, a20, a21, a22, a23)
    GEMM_ROW(3, a30, a31, a32, a33)
    __syncthreads();
  }

#define GEMM_EPI(MT, NT, ACC)                                                 \
  { int col = col0 + cw * 64 + NT * 16 + n;                                   \
    float bz = bias ? bias[col] : 0.f;                                        \
    U8f u; u.v = ACC;                                                         \
    _Pragma("unroll")                                                         \
    for (int vv = 0; vv < 8; ++vv) {                                          \
      int r = row0 + rw * 64 + MT * 16 + hlf * 8 + vv;                        \
      size_t o = (size_t)r * D + col;                                         \
      float val = u.f[vv] + bz;                                               \
      if (res) val += (float)res[o];                                          \
      if (outF) outF[o] = val;                                                \
      if (outH) outH[o] = (half_t)val;                                        \
    } }
  GEMM_EPI(0, 0, a00) GEMM_EPI(0, 1, a01) GEMM_EPI(0, 2, a02) GEMM_EPI(0, 3, a03)
  GEMM_EPI(1, 0, a10) GEMM_EPI(1, 1, a11) GEMM_EPI(1, 2, a12) GEMM_EPI(1, 3, a13)
  GEMM_EPI(2, 0, a20) GEMM_EPI(2, 1, a21) GEMM_EPI(2, 2, a22) GEMM_EPI(2, 3, a23)
  GEMM_EPI(3, 0, a30) GEMM_EPI(3, 1, a31) GEMM_EPI(3, 2, a32) GEMM_EPI(3, 3, a33)
}

// ---------------- GroupNorm(32 groups of 16ch) + ReLU, f32 in -> f16 out -------------
__global__ __launch_bounds__(256) void k_gn_relu(
    const float* __restrict__ X, const float* __restrict__ g,
    const float* __restrict__ be, half_t* __restrict__ out, int M)
{
  int i = blockIdx.x * 256 + threadIdx.x;
  if (i >= M * 32) return;
  int pt = i >> 5, gr = i & 31;
  const float* x = X + (size_t)pt * 512 + gr * 16;
  float v[16];
  #pragma unroll
  for (int j = 0; j < 16; j += 4) {
    float4 a = *(const float4*)(x + j);
    v[j] = a.x; v[j + 1] = a.y; v[j + 2] = a.z; v[j + 3] = a.w;
  }
  float m = 0.f;
  #pragma unroll
  for (int j = 0; j < 16; ++j) m += v[j];
  m *= 0.0625f;
  float var = 0.f;
  #pragma unroll
  for (int j = 0; j < 16; ++j) { float d = v[j] - m; var += d * d; }
  var *= 0.0625f;
  float inv = rsqrtf(var + 1e-5f);
  H8 o0, o1;
  #pragma unroll
  for (int j = 0; j < 8; ++j)
    o0.h[j] = (half_t)fmaxf((v[j] - m) * inv * g[gr * 16 + j] + be[gr * 16 + j], 0.f);
  #pragma unroll
  for (int j = 0; j < 8; ++j)
    o1.h[j] = (half_t)fmaxf((v[8 + j] - m) * inv * g[gr * 16 + 8 + j] + be[gr * 16 + 8 + j], 0.f);
  half_t* op = out + (size_t)pt * 512 + gr * 16;
  *(uint4*)op = o0.q;
  *(uint4*)(op + 8) = o1.q;
}

// ---------------- small-D GEMM (D=6 or 3): WT is [D,K] ----------------
__global__ void k_small_gemm(const half_t* __restrict__ X, const half_t* __restrict__ WT,
                             const float* __restrict__ bias, float* __restrict__ out,
                             int M, int K, int D)
{
  int i = blockIdx.x * 256 + threadIdx.x;
  if (i >= M * D) return;
  int row = i / D, d = i % D;
  const half_t* xp = X + (size_t)row * K;
  const half_t* wp = WT + (size_t)d * K;
  float acc = 0.f;
  for (int k = 0; k < K; k += 8) {
    H8 a, b; a.q = *(const uint4*)(xp + k); b.q = *(const uint4*)(wp + k);
    #pragma unroll
    for (int j = 0; j < 8; ++j) acc += (float)a.h[j] * (float)b.h[j];
  }
  out[i] = acc + bias[d];
}

// ---------------- xyz kernels ----------------
__global__ void k_build_xyz(const float* __restrict__ xyz, const float* __restrict__ split,
                            float* __restrict__ xyz2, int N2, float scale, int total)
{
  int i = blockIdx.x * 256 + threadIdx.x;
  if (i >= total) return;
  int m = i / 3, k = i % 3;
  int b = m / N2, n = m % N2;
  int N = N2 >> 1;
  int src = b * N + (n < N ? n : n - N);
  int col = (n < N) ? k : k + 3;
  xyz2[i] = xyz[src * 3 + k] + split[src * 6 + col] * scale;
}

__global__ void k_add_scaled(float* __restrict__ xyz, const float* __restrict__ off, float s, int total) {
  int i = blockIdx.x * 256 + threadIdx.x;
  if (i < total) xyz[i] += off[i] * s;
}

__global__ void k_repulsion(const float* __restrict__ xin, float* __restrict__ xout,
                            int N2, float mind, int total)
{
  int i = blockIdx.x * 256 + threadIdx.x;
  if (i >= total) return;
  int b = i / N2, p = i % N2;
  const float* base = xin + (size_t)b * N2 * 3;
  float px = base[p * 3], py = base[p * 3 + 1], pz = base[p * 3 + 2];
  float fx = 0.f, fy = 0.f, fz = 0.f;
  for (int j = 0; j < N2; ++j) {
    float dx = px - base[j * 3], dy = py - base[j * 3 + 1], dz = pz - base[j * 3 + 2];
    float dist = sqrtf(dx * dx + dy * dy + dz * dz) + (j == p ? 1e6f : 0.f);
    if (dist < mind) {
      float inv = 1.f / (dist + 1e-8f);
      fx += dx * inv; fy += dy * inv; fz += dz * inv;
    }
  }
  xout[i * 3 + 0] = px + fx * 0.01f;
  xout[i * 3 + 1] = py + fy * 0.01f;
  xout[i * 3 + 2] = pz + fz * 0.01f;
}

// ---------------- flash attention: 16 q-rows per WG, 8 waves x 16 keys/iter, 2048 keys ----
__global__ __launch_bounds__(256, 1) void k_attn(
    const half_t* __restrict__ Q, const half_t* __restrict__ Kv,
    const half_t* __restrict__ V, half_t* __restrict__ Out,
    int N2, float scale)
{
  __shared__ float Sb[8][16][16];
  __shared__ half_t Pb[16][128];
  __shared__ float m_s[16], l_s[16], al_s[16];
  __shared__ float bmax[8][16], bsum[8][16];

  int tid = threadIdx.x;
  int w = tid >> 5, lane = tid & 31;
  int n = lane & 15, h = lane >> 4;
  int row0 = blockIdx.x * 16;
  int batch = row0 / N2;
  size_t kv = (size_t)batch * 2048 * 512;

  if (tid < 16) { m_s[tid] = -1e30f; l_s[tid] = 0.f; }

  // preload all Q A-fragments (16 k-slices of 32)
  U16h qa[16];
  const half_t* qrow = Q + (size_t)(row0 + n) * 512;
  #pragma unroll
  for (int s = 0; s < 16; ++s)
    #pragma unroll
    for (int j = 0; j < 8; ++j) {
      int kd = s * 32 + (h << 4) + (j << 1);
      qa[s].u[j] = *(const u32*)(qrow + kd);
    }

  v8f o0 = {}, o1 = {}, o2 = {}, o3 = {};

  __syncthreads();

  const u16* Vu = (const u16*)(V + kv);

  for (int it = 0; it < 16; ++it) {
    int keyb = it * 128 + w * 16;
    v8f sc = {};
    const half_t* kp0 = Kv + kv + (size_t)(keyb + n) * 512 + (h << 4);
    #pragma unroll
    for (int s = 0; s < 16; ++s) {
      U16h bk;
      const half_t* kp = kp0 + s * 32;
      #pragma unroll
      for (int j = 0; j < 8; ++j) bk.u[j] = *(const u32*)(kp + 2 * j);
      sc = wmma_f32_f16(qa[s].v, bk.v, sc);
    }
    {
      U8f u; u.v = sc;
      #pragma unroll
      for (int v = 0; v < 8; ++v) Sb[w][h * 8 + v][n] = u.f[v] * scale;
    }
    __syncthreads();

    if (lane < 16) {
      float mx = -1e30f;
      #pragma unroll
      for (int j = 0; j < 16; ++j) mx = fmaxf(mx, Sb[w][lane][j]);
      bmax[w][lane] = mx;
    }
    __syncthreads();

    if (w == 0 && lane < 16) {
      float nm = m_s[lane];
      #pragma unroll
      for (int ww = 0; ww < 8; ++ww) nm = fmaxf(nm, bmax[ww][lane]);
      al_s[lane] = __expf(m_s[lane] - nm);
      m_s[lane] = nm;
    }
    __syncthreads();

    if (lane < 16) {
      float nm = m_s[lane]; float sum = 0.f;
      #pragma unroll
      for (int j = 0; j < 16; ++j) {
        float p = __expf(Sb[w][lane][j] - nm);
        Pb[lane][w * 16 + j] = (half_t)p;
        sum += p;
      }
      bsum[w][lane] = sum;
    }
    __syncthreads();

    if (w == 0 && lane < 16) {
      float l = l_s[lane] * al_s[lane];
      #pragma unroll
      for (int ww = 0; ww < 8; ++ww) l += bsum[ww][lane];
      l_s[lane] = l;
    }

    v8f alv;
    {
      U8f ua;
      #pragma unroll
      for (int v = 0; v < 8; ++v) ua.f[v] = al_s[h * 8 + v];
      alv = ua.v;
    }
    o0 *= alv; o1 *= alv; o2 *= alv; o3 *= alv;

    U16h pa[4];
    #pragma unroll
    for (int s = 0; s < 4; ++s)
      #pragma unroll
      for (int j = 0; j < 8; ++j) {
        int kk = s * 32 + (h << 4) + (j << 1);
        pa[s].u[j] = *(const u32*)&Pb[n][kk];
      }
#define ATTN_PV(T, O)                                                         \
    { int col = w * 64 + T * 16 + n;                                          \
      _Pragma("unroll")                                                       \
      for (int s = 0; s < 4; ++s) {                                           \
        U16h bv;                                                              \
        _Pragma("unroll")                                                     \
        for (int j = 0; j < 8; ++j) {                                         \
          int key = it * 128 + s * 32 + (h << 4) + 2 * j;                     \
          u32 lo = Vu[(size_t)key * 512 + col];                               \
          u32 hi = Vu[(size_t)(key + 1) * 512 + col];                         \
          bv.u[j] = lo | (hi << 16);                                          \
        }                                                                     \
        O = wmma_f32_f16(pa[s].v, bv.v, O);                                   \
      } }
    ATTN_PV(0, o0) ATTN_PV(1, o1) ATTN_PV(2, o2) ATTN_PV(3, o3)
    __syncthreads();
  }

  float li[8];
  #pragma unroll
  for (int v = 0; v < 8; ++v) li[v] = 1.0f / l_s[h * 8 + v];
#define ATTN_STO(T, O)                                                        \
  { U8f u; u.v = O;                                                           \
    _Pragma("unroll")                                                         \
    for (int v = 0; v < 8; ++v)                                               \
      Out[(size_t)(row0 + h * 8 + v) * 512 + w * 64 + T * 16 + n] =           \
          (half_t)(u.f[v] * li[v]); }
  ATTN_STO(0, o0) ATTN_STO(1, o1) ATTN_STO(2, o2) ATTN_STO(3, o3)
}

// ======================= host =======================
extern "C" void kernel_launch(void* const* d_in, const int* in_sizes, int n_in,
                              void* d_out, int out_size, void* d_ws, size_t ws_size,
                              hipStream_t stream) {
  (void)out_size; (void)ws_size;
  const int B = 2;
  // input layout auto-detect: insertion order (partial_xyz first) vs fully pytree-sorted
  int pbase, iPF, iGF;
  if (in_sizes[0] == 12288) { iPF = 1; iGF = 2; pbase = 3; }
  else { iGF = 0; pbase = 1; iPF = n_in - 2; }
  auto F = [&](int i) { return (const float*)d_in[i]; };
  const float* partial_feat = F(iPF);
  const float* global_feat = F(iGF);
  const float* init_points = F(pbase + 30);
  // pytree-sorted leaf order inside params:
  auto coords_p = [&](int i, int j) { return F(pbase + 6 * i + j); };        // c1W,c1b,c2W,c2b,cg1be,cg1g
  auto props_p  = [&](int i, int j) { return F(pbase + 31 + 8 * i + j); };   // kW,kb,oW,ob,qW,qb,vW,vb
  auto stages_p = [&](int i, int j) { return F(pbase + 71 + 14 * i + j); };  // r1W,r1b,r2W,r2b,rg1be,rg1g,rg2be,rg2g,s1W,s1b,s2W,s2b,sg1be,sg1g

  char* base = (char*)d_ws; size_t off = 0;
  auto alloc = [&](size_t bytes) -> void* {
    void* p = base + off; off += (bytes + 255) & ~(size_t)255; return p;
  };

  half_t *wt_r1[5], *wt_r2[5], *wt_s1[5], *wt_s2T[5], *wt_q[5], *wt_k[5],
         *wt_v[5], *wt_o[5], *wt_c1[5], *wt_c2T[5];
  for (int i = 0; i < 5; ++i) {
    wt_r1[i]  = (half_t*)alloc((size_t)1024 * 512 * 2);
    wt_r2[i]  = (half_t*)alloc((size_t)512 * 512 * 2);
    wt_s1[i]  = (half_t*)alloc((size_t)1024 * 512 * 2);
    wt_s2T[i] = (half_t*)alloc((size_t)6 * 512 * 2);
    wt_q[i]   = (half_t*)alloc((size_t)512 * 512 * 2);
    wt_k[i]   = (half_t*)alloc((size_t)512 * 512 * 2);
    wt_v[i]   = (half_t*)alloc((size_t)512 * 512 * 2);
    wt_o[i]   = (half_t*)alloc((size_t)512 * 512 * 2);
    wt_c1[i]  = (half_t*)alloc((size_t)512 * 512 * 2);
    wt_c2T[i] = (half_t*)alloc((size_t)3 * 512 * 2);
  }
  half_t* pf16   = (half_t*)alloc((size_t)4096 * 512 * 2);
  half_t* featA  = (half_t*)alloc((size_t)16384 * 512 * 2);
  half_t* featB  = (half_t*)alloc((size_t)16384 * 512 * 2);
  half_t* comb   = (half_t*)alloc((size_t)8192 * 1024 * 2);
  float*  tmp    = (float*)alloc((size_t)16384 * 512 * 4);
  half_t* h16    = (half_t*)alloc((size_t)8192 * 512 * 2);
  half_t* fr16   = (half_t*)alloc((size_t)8192 * 512 * 2);
  half_t* s16    = (half_t*)alloc((size_t)8192 * 512 * 2);
  half_t* q16    = (half_t*)alloc((size_t)16384 * 512 * 2);
  half_t* k16    = (half_t*)alloc((size_t)4096 * 512 * 2);
  half_t* v16buf = (half_t*)alloc((size_t)4096 * 512 * 2);
  half_t* ctx16  = (half_t*)alloc((size_t)16384 * 512 * 2);
  float*  splitf = (float*)alloc((size_t)8192 * 6 * 4);
  float*  offf   = (float*)alloc((size_t)16384 * 3 * 4);
  float*  xyzA   = (float*)alloc((size_t)16384 * 3 * 4);
  float*  xyzB   = (float*)alloc((size_t)16384 * 3 * 4);
  half_t* cact16 = comb;  // comb dead by the time cact is produced

  auto nb = [](long long t) { return (unsigned)((t + 255) / 256); };
  auto cvt = [&](const float* src, half_t* dst, int nelem) {
    k_cvt_f16<<<nb(nelem), 256, 0, stream>>>(src, dst, nelem);
  };

  for (int i = 0; i < 5; ++i) {
    cvt(stages_p(i, 0), wt_r1[i], 1024 * 512);
    cvt(stages_p(i, 2), wt_r2[i], 512 * 512);
    cvt(stages_p(i, 8), wt_s1[i], 1024 * 512);
    k_cvt_f16_T<<<nb(512 * 6), 256, 0, stream>>>(stages_p(i, 10), wt_s2T[i], 512, 6);
    cvt(props_p(i, 4), wt_q[i], 512 * 512);
    cvt(props_p(i, 0), wt_k[i], 512 * 512);
    cvt(props_p(i, 6), wt_v[i], 512 * 512);
    cvt(props_p(i, 2), wt_o[i], 512 * 512);
    cvt(coords_p(i, 0), wt_c1[i], 512 * 512);
    k_cvt_f16_T<<<nb(512 * 3), 256, 0, stream>>>(coords_p(i, 2), wt_c2T[i], 512, 3);
  }
  cvt(partial_feat, pf16, B * 2048 * 512);
  k_init_feat<<<nb((long long)B * 256 * 64), 256, 0, stream>>>(global_feat, featA, B * 256, 256);
  k_init_xyz<<<nb(B * 256 * 3), 256, 0, stream>>>(init_points, xyzA, B * 256 * 3);

  half_t* featP = featA;
  half_t* featQ = featB;
  const float inv_sqrt_d = 0.04419417382415922f;  // 1/sqrt(512)

  for (int i = 0; i < 5; ++i) {
    int N = 256 << i, N2 = N * 2;
    int Mi = B * N, Mo = B * N2;

    k_build_comb<<<nb((long long)Mi * 128), 256, 0, stream>>>(featP, global_feat, comb, Mi, N);

    k_gemm<<<dim3(4, Mi / 128), 128, 0, stream>>>(comb, wt_r1[i], stages_p(i, 1), tmp, nullptr, nullptr, Mi, 1024, 512);
    k_gn_relu<<<nb((long long)Mi * 32), 256, 0, stream>>>(tmp, stages_p(i, 5), stages_p(i, 4), h16, Mi);
    k_gemm<<<dim3(4, Mi / 128), 128, 0, stream>>>(h16, wt_r2[i], stages_p(i, 3), tmp, nullptr, nullptr, Mi, 512, 512);
    k_gn_relu<<<nb((long long)Mi * 32), 256, 0, stream>>>(tmp, stages_p(i, 7), stages_p(i, 6), fr16, Mi);
    k_gemm<<<dim3(4, Mi / 128), 128, 0, stream>>>(comb, wt_s1[i], stages_p(i, 9), tmp, nullptr, nullptr, Mi, 1024, 512);
    k_gn_relu<<<nb((long long)Mi * 32), 256, 0, stream>>>(tmp, stages_p(i, 13), stages_p(i, 12), s16, Mi);
    k_small_gemm<<<nb((long long)Mi * 6), 256, 0, stream>>>(s16, wt_s2T[i], stages_p(i, 11), splitf, Mi, 512, 6);

    float sc1 = 0.05f / (float)(i + 1);
    k_build_xyz<<<nb((long long)Mo * 3), 256, 0, stream>>>(xyzA, splitf, xyzB, N2, sc1, Mo * 3);
    k_dup_feat<<<nb((long long)Mo * 64), 256, 0, stream>>>(fr16, featQ, Mo, N2);

    k_gemm<<<dim3(4, Mo / 128), 128, 0, stream>>>(featQ, wt_q[i], props_p(i, 5), nullptr, q16, nullptr, Mo, 512, 512);
    k_gemm<<<dim3(4, 4096 / 128), 128, 0, stream>>>(pf16, wt_k[i], props_p(i, 1), nullptr, k16, nullptr, 4096, 512, 512);
    k_gemm<<<dim3(4, 4096 / 128), 128, 0, stream>>>(pf16, wt_v[i], props_p(i, 7), nullptr, v16buf, nullptr, 4096, 512, 512);
    k_attn<<<Mo / 16, 256, 0, stream>>>(q16, k16, v16buf, ctx16, N2, inv_sqrt_d);
    k_gemm<<<dim3(4, Mo / 128), 128, 0, stream>>>(ctx16, wt_o[i], props_p(i, 3), nullptr, featQ, featQ, Mo, 512, 512);

    k_gemm<<<dim3(4, Mo / 128), 128, 0, stream>>>(featQ, wt_c1[i], coords_p(i, 1), tmp, nullptr, nullptr, Mo, 512, 512);
    k_gn_relu<<<nb((long long)Mo * 32), 256, 0, stream>>>(tmp, coords_p(i, 5), coords_p(i, 4), cact16, Mo);
    k_small_gemm<<<nb((long long)Mo * 3), 256, 0, stream>>>(cact16, wt_c2T[i], coords_p(i, 3), offf, Mo, 512, 3);
    k_add_scaled<<<nb((long long)Mo * 3), 256, 0, stream>>>(xyzB, offf, 0.1f / (float)(i + 1), Mo * 3);

    if (i < 4) {
      float mind = 0.01f / (float)(1 << i);
      k_repulsion<<<nb(Mo), 256, 0, stream>>>(xyzB, xyzA, N2, mind, Mo);
    } else {
      (void)hipMemcpyAsync(d_out, xyzB, (size_t)Mo * 3 * sizeof(float), hipMemcpyDeviceToDevice, stream);
    }
    half_t* t = featP; featP = featQ; featQ = t;
  }
}
